// CausalSelfAttention_3582002725214
// MI455X (gfx1250) — matrix operand order repaired and checked by reading, not compiled, not run on hardware
//
#include <hip/hip_runtime.h>
#include <math.h>

// ---------------------------------------------------------------------------
// CDNA5 (gfx1250) causal self-attention: bf16 WMMA + async global->LDS copies.
//   B=2, T=4096, E=768, H=12, D=64
// ---------------------------------------------------------------------------

typedef __attribute__((ext_vector_type(16))) __bf16 v16bf;
typedef __attribute__((ext_vector_type(8)))  __bf16 v8bf;
typedef __attribute__((ext_vector_type(8)))  float  v8f;
typedef int v4i __attribute__((vector_size(16)));   // matches builtin param type

#define AS1 __attribute__((address_space(1)))
#define AS3 __attribute__((address_space(3)))

#define BB 2
#define TT 4096
#define EE 768
#define HH 12
#define DD 64
#define MTOT (BB * TT)      // 8192
#define N3E  (3 * EE)       // 2304

// ---- async global->LDS (gfx1250) with guarded fallback --------------------
#if defined(__has_builtin)
#if __has_builtin(__builtin_amdgcn_global_load_async_to_lds_b128)
#define HAVE_ASYNC_LDS 1
#endif
#endif
#ifndef HAVE_ASYNC_LDS
#define HAVE_ASYNC_LDS 0
#endif

#if HAVE_ASYNC_LDS
#if __has_builtin(__builtin_amdgcn_s_wait_asynccnt)
#define ASYNC_WAIT(n) __builtin_amdgcn_s_wait_asynccnt(n)
#else
#define ASYNC_WAIT(n) asm volatile("s_wait_asynccnt %0" ::"i"(n))
#endif
#else
#define ASYNC_WAIT(n) do {} while (0)
#endif

// copy 16 bytes (8 bf16) global -> LDS
__device__ __forceinline__ void cp16(const unsigned short* g, unsigned short* l) {
#if HAVE_ASYNC_LDS
  AS1 void* gp = (AS1 void*)g;   // addrspacecast to global
  AS3 void* lp = (AS3 void*)l;   // addrspacecast to LDS
  __builtin_amdgcn_global_load_async_to_lds_b128((AS1 v4i*)gp, (AS3 v4i*)lp, 0, 0);
#else
  *(uint4*)l = *(const uint4*)g;
#endif
}

__device__ __forceinline__ unsigned short f32_to_bf16(float f) {
  unsigned int u = __float_as_uint(f);
  unsigned int r = u + 0x7FFFu + ((u >> 16) & 1u);   // round-to-nearest-even
  return (unsigned short)(r >> 16);
}

// ---------------------------------------------------------------------------
// fp32 -> bf16 conversion (straight and transposing)
// ---------------------------------------------------------------------------
__global__ void cvt_bf16_kernel(const float* __restrict__ src,
                                unsigned short* __restrict__ dst, int n) {
  int i = blockIdx.x * blockDim.x + threadIdx.x;
  if (i < n) dst[i] = f32_to_bf16(src[i]);
}

// src [K][N] fp32 -> dst [N][K] bf16
__global__ void cvt_transpose_bf16_kernel(const float* __restrict__ src,
                                          unsigned short* __restrict__ dst,
                                          int K, int N) {
  int i = blockIdx.x * blockDim.x + threadIdx.x;
  if (i < K * N) {
    int k = i / N, n = i - k * N;
    dst[(size_t)n * K + k] = f32_to_bf16(src[i]);
  }
}

// ---------------------------------------------------------------------------
// Tiled bf16 GEMM: C = A[M,K] * Bt[N,K]^T + bias  (Bt pre-transposed).
//   MODE 0: QKV epilogue -> Q/K [B,H,T,64] bf16, V transposed [B,H,64,T] bf16
//   MODE 1: fp32 epilogue -> out0 (float [M,N])
// WG = 128 threads (4 waves). WG tile = 64x64, double-buffered async staging.
// ---------------------------------------------------------------------------
template <int MODE>
__global__ __launch_bounds__(128)
void gemm_bf16_kernel(const unsigned short* __restrict__ A,   // [M][K]
                      const unsigned short* __restrict__ Bt,  // [N][K]
                      const float* __restrict__ bias,
                      void* __restrict__ out0, void* __restrict__ out1,
                      void* __restrict__ out2,
                      int M, int N, int K) {
  __shared__ __attribute__((aligned(32))) unsigned short lds_a[2][64 * 32]; // [m][k]
  __shared__ __attribute__((aligned(32))) unsigned short lds_b[2][64 * 32]; // [n][k]

  const int tid  = threadIdx.x;
  const int wid  = tid >> 5;
  const int lane = tid & 31;
  const int l16  = lane >> 4;
  const int lm   = lane & 15;
  const int m0   = blockIdx.y * 64;
  const int n0   = blockIdx.x * 64;
  const int srow = tid >> 1, shalf = tid & 1;   // staging: row 0..63, 16-elem half

  v8f acc[4];
#pragma unroll
  for (int i = 0; i < 4; ++i)
#pragma unroll
    for (int r = 0; r < 8; ++r) acc[i][r] = 0.f;

  // stage one 64x32 A tile + 64x32 B tile into buffer `buf` (4 cp16 / thread)
  auto stage = [&](int kk, int buf) {
    {
      const unsigned short* g = A + (size_t)(m0 + srow) * K + kk + shalf * 16;
      unsigned short* l = &lds_a[buf][srow * 32 + shalf * 16];
      cp16(g, l);
      cp16(g + 8, l + 8);
    }
    {
      const unsigned short* g = Bt + (size_t)(n0 + srow) * K + kk + shalf * 16;
      unsigned short* l = &lds_b[buf][srow * 32 + shalf * 16];
      cp16(g, l);
      cp16(g + 8, l + 8);
    }
  };

  const int steps = K >> 5;   // K/32
  stage(0, 0);
  for (int s = 0; s < steps; ++s) {
    const int buf = s & 1;
    if (s + 1 < steps) {
      stage((s + 1) * 32, (s + 1) & 1);  // prefetch next tile into other buffer
      ASYNC_WAIT(4);                     // newest batch (4 ops) may stay in flight
    } else {
      ASYNC_WAIT(0);
    }
    __syncthreads();

    // B fragment: e<8 -> K = l16*8+e ; e>=8 -> K = 16 + l16*8 + (e-8), N = lm
    union { v16bf v; v8bf h[2]; } bu;
    const unsigned short* bp = &lds_b[buf][(wid * 16 + lm) * 32 + l16 * 8];
    bu.h[0] = *(const v8bf*)bp;
    bu.h[1] = *(const v8bf*)(bp + 16);
    v16bf bfrag = bu.v;
#pragma unroll
    for (int mt = 0; mt < 4; ++mt) {
      // A fragment: e<8 -> K = l16*8+e ; e>=8 -> K = 16 + l16*8 + (e-8)
      union { v16bf v; v8bf h[2]; } a;
      const unsigned short* ap = &lds_a[buf][(mt * 16 + lm) * 32 + l16 * 8];
      a.h[0] = *(const v8bf*)ap;
      a.h[1] = *(const v8bf*)(ap + 16);
      acc[mt] = __builtin_amdgcn_wmma_f32_16x16x32_bf16(
          false, a.v, false, bfrag, (short)0, acc[mt], false, false);
    }
    __syncthreads();   // release buffer `buf` for the prefetch two steps ahead
  }

  // ---- epilogue ----
  const int j = n0 + wid * 16 + lm;    // global column
  const float bj = bias[j];
#pragma unroll
  for (int mt = 0; mt < 4; ++mt) {
#pragma unroll
    for (int r = 0; r < 8; ++r) {
      const int m = m0 + mt * 16 + r + 8 * l16;   // C layout: row = r + 8*l16
      const float v = acc[mt][r] + bj;
      if (MODE == 0) {
        const int which = j / EE;              // 0=Q 1=K 2=V
        const int jj = j - which * EE;
        const int h = jj >> 6, d = jj & 63;
        const int b = m >> 12, t = m & 4095;
        const int bh = b * HH + h;
        if (which == 2) {   // V stored transposed: [bh][d][t]
          ((unsigned short*)out2)[((size_t)(bh * DD + d)) * TT + t] = f32_to_bf16(v);
        } else {
          unsigned short* dst = (which == 0) ? (unsigned short*)out0
                                             : (unsigned short*)out1;
          dst[((size_t)bh * TT + t) * DD + d] = f32_to_bf16(v);
        }
      } else {
        ((float*)out0)[(size_t)m * N + j] = v;
      }
    }
  }
}

// ---------------------------------------------------------------------------
// Flash attention (causal). Grid: x = q-tile (T/64 = 64), y = b*H+h (24).
// WG = 128 threads = 4 waves. Wave w owns query rows [q0+16w, q0+16w+16).
// K-tile [key][d] and pre-transposed V-tile [d][key] staged with async copies,
// double-buffered across the key-tile loop.
// ---------------------------------------------------------------------------
__global__ __launch_bounds__(128)
void attn_kernel(const unsigned short* __restrict__ Qb,   // [bh][t][64]
                 const unsigned short* __restrict__ Kb,   // [bh][t][64]
                 const unsigned short* __restrict__ Vt,   // [bh][64][t]
                 unsigned short* __restrict__ Yb) {
  __shared__ __attribute__((aligned(32))) unsigned short lds_k[2][64 * 64]; // [key][d]
  __shared__ __attribute__((aligned(32))) unsigned short lds_v[2][64 * 64]; // [d][key]
  __shared__ __attribute__((aligned(32))) unsigned short lds_p[4 * 16 * 64];

  const int tid  = threadIdx.x;
  const int wid  = tid >> 5;
  const int lane = tid & 31;
  const int l16  = lane >> 4;
  const int lm   = lane & 15;
  const int qt = blockIdx.x;
  const int bh = blockIdx.y;
  const int b  = bh / HH, h = bh % HH;
  const int q0 = qt * 64;
  const size_t head_base = (size_t)bh * TT * DD;
  const int srow = tid >> 1, shalf = tid & 1;

  // stage K tile (straight) + V tile (already transposed) : 8 cp16 / thread
  auto stage_kv = [&](int k0, int buf) {
    {
      const unsigned short* g = Kb + head_base + (size_t)(k0 + srow) * DD + shalf * 32;
      unsigned short* l = &lds_k[buf][srow * 64 + shalf * 32];
#pragma unroll
      for (int i = 0; i < 4; ++i) cp16(g + i * 8, l + i * 8);
    }
    {
      const unsigned short* g = Vt + head_base + (size_t)srow * TT + k0 + shalf * 32;
      unsigned short* l = &lds_v[buf][srow * 64 + shalf * 32];
#pragma unroll
      for (int i = 0; i < 4; ++i) cp16(g + i * 8, l + i * 8);
    }
  };

  // Q A-fragments (held in registers for the whole kernel)
  const int qrow = q0 + wid * 16 + lm;
  union { v16bf v; v8bf h2[2]; } qfrag[2];
#pragma unroll
  for (int dd = 0; dd < 2; ++dd) {
    const unsigned short* qp = Qb + head_base + (size_t)qrow * DD + dd * 32 + l16 * 8;
    qfrag[dd].h2[0] = *(const v8bf*)qp;
    qfrag[dd].h2[1] = *(const v8bf*)(qp + 16);
  }

  float m_state[8], l_state[8];
  v8f o[4];
#pragma unroll
  for (int r = 0; r < 8; ++r) { m_state[r] = -INFINITY; l_state[r] = 0.f; }
#pragma unroll
  for (int n = 0; n < 4; ++n)
#pragma unroll
    for (int r = 0; r < 8; ++r) o[n][r] = 0.f;

  const float sm_scale = 0.125f;  // 1/sqrt(64)

  stage_kv(0, 0);
  for (int kt = 0; kt <= qt; ++kt) {
    const int k0 = kt * 64;
    const int buf = kt & 1;
    if (kt < qt) {
      stage_kv((kt + 1) * 64, (kt + 1) & 1);  // prefetch next key tile
      ASYNC_WAIT(8);
    } else {
      ASYNC_WAIT(0);
    }
    __syncthreads();

    // ---- S = Q * K^T : 4 key sub-tiles x 2 K-steps = 8 WMMAs ----
    v8f s[4];
#pragma unroll
    for (int n = 0; n < 4; ++n)
#pragma unroll
      for (int r = 0; r < 8; ++r) s[n][r] = 0.f;
#pragma unroll
    for (int n = 0; n < 4; ++n) {
#pragma unroll
      for (int dd = 0; dd < 2; ++dd) {
        union { v16bf v; v8bf h2[2]; } ku;
        const unsigned short* kp = &lds_k[buf][(n * 16 + lm) * 64 + dd * 32 + l16 * 8];
        ku.h2[0] = *(const v8bf*)kp;
        ku.h2[1] = *(const v8bf*)(kp + 16);
        v16bf kf = ku.v;
        s[n] = __builtin_amdgcn_wmma_f32_16x16x32_bf16(
            false, qfrag[dd].v, false, kf, (short)0, s[n], false, false);
      }
    }

    // ---- scale + causal mask ----
#pragma unroll
    for (int n = 0; n < 4; ++n) {
      const int key = k0 + n * 16 + lm;
#pragma unroll
      for (int r = 0; r < 8; ++r) {
        const int q = q0 + wid * 16 + r + 8 * l16;
        const float val = s[n][r] * sm_scale;
        s[n][r] = (key <= q) ? val : -INFINITY;
      }
    }

    // ---- online softmax (row = r + 8*l16; reduce across 16-lane groups) ----
#pragma unroll
    for (int r = 0; r < 8; ++r) {
      float mx = fmaxf(fmaxf(s[0][r], s[1][r]), fmaxf(s[2][r], s[3][r]));
      mx = fmaxf(mx, __shfl_xor(mx, 1, 32));
      mx = fmaxf(mx, __shfl_xor(mx, 2, 32));
      mx = fmaxf(mx, __shfl_xor(mx, 4, 32));
      mx = fmaxf(mx, __shfl_xor(mx, 8, 32));
      const float m_new = fmaxf(m_state[r], mx);
      const float scale = __expf(m_state[r] - m_new);
      float sum = 0.f;
#pragma unroll
      for (int n = 0; n < 4; ++n) {
        const float p = __expf(s[n][r] - m_new);
        s[n][r] = p;
        sum += p;
      }
      sum += __shfl_xor(sum, 1, 32);
      sum += __shfl_xor(sum, 2, 32);
      sum += __shfl_xor(sum, 4, 32);
      sum += __shfl_xor(sum, 8, 32);
      l_state[r] = l_state[r] * scale + sum;
      m_state[r] = m_new;
#pragma unroll
      for (int n = 0; n < 4; ++n) o[n][r] *= scale;
    }

    // ---- reshape P (C layout -> A layout) through wave-private LDS ----
    unsigned short* pbuf = lds_p + wid * (16 * 64);
#pragma unroll
    for (int n = 0; n < 4; ++n)
#pragma unroll
      for (int r = 0; r < 8; ++r)
        pbuf[(r + 8 * l16) * 64 + n * 16 + lm] = f32_to_bf16(s[n][r]);
    __syncthreads();  // order cross-lane LDS writes before A-fragment reads

    // ---- O += P * V : 2 K-steps x 4 D sub-tiles = 8 WMMAs ----
#pragma unroll
    for (int kk = 0; kk < 2; ++kk) {
      union { v16bf v; v8bf h2[2]; } pf;
      const unsigned short* pp = pbuf + lm * 64 + kk * 32 + l16 * 8;
      pf.h2[0] = *(const v8bf*)pp;
      pf.h2[1] = *(const v8bf*)(pp + 16);
#pragma unroll
      for (int n = 0; n < 4; ++n) {
        union { v16bf v; v8bf h2[2]; } vu;
        const unsigned short* vp = &lds_v[buf][(n * 16 + lm) * 64 + kk * 32 + l16 * 8];
        vu.h2[0] = *(const v8bf*)vp;
        vu.h2[1] = *(const v8bf*)(vp + 16);
        v16bf vf = vu.v;
        o[n] = __builtin_amdgcn_wmma_f32_16x16x32_bf16(
            false, pf.v, false, vf, (short)0, o[n], false, false);
      }
    }
    __syncthreads();  // release K/V buffer `buf` and lds_p for reuse
  }

  // ---- normalize + write Y[b, t, h*64 + d] (bf16) ----
#pragma unroll
  for (int n = 0; n < 4; ++n) {
#pragma unroll
    for (int r = 0; r < 8; ++r) {
      const int q = q0 + wid * 16 + r + 8 * l16;
      const int d = n * 16 + lm;
      const float val = o[n][r] / l_state[r];
      Yb[((size_t)(b * TT + q)) * EE + h * DD + d] = f32_to_bf16(val);
    }
  }
}

// ---------------------------------------------------------------------------
// Launcher
// ---------------------------------------------------------------------------
extern "C" void kernel_launch(void* const* d_in, const int* in_sizes, int n_in,
                              void* d_out, int out_size, void* d_ws, size_t ws_size,
                              hipStream_t stream) {
  const float* x      = (const float*)d_in[0];
  const float* W_attn = (const float*)d_in[1];
  const float* b_attn = (const float*)d_in[2];
  const float* W_proj = (const float*)d_in[3];
  const float* b_proj = (const float*)d_in[4];
  float* out = (float*)d_out;

  // workspace carve-up (all 16B aligned)
  char* ws = (char*)d_ws;
  unsigned short* xb  = (unsigned short*)ws; ws += (size_t)MTOT * EE * 2;
  unsigned short* Wat = (unsigned short*)ws; ws += (size_t)EE * N3E * 2;  // [2304][768]
  unsigned short* Wpt = (unsigned short*)ws; ws += (size_t)EE * EE * 2;   // [768][768]
  unsigned short* Qb  = (unsigned short*)ws; ws += (size_t)BB * HH * TT * DD * 2;
  unsigned short* Kb  = (unsigned short*)ws; ws += (size_t)BB * HH * TT * DD * 2;
  unsigned short* Vt  = (unsigned short*)ws; ws += (size_t)BB * HH * TT * DD * 2;
  unsigned short* Yb  = (unsigned short*)ws; ws += (size_t)MTOT * EE * 2;

  // 1) fp32 -> bf16 conversions (weights transposed to [N][K])
  {
    int n1 = MTOT * EE;
    cvt_bf16_kernel<<<(n1 + 255) / 256, 256, 0, stream>>>(x, xb, n1);
    int n2 = EE * N3E;
    cvt_transpose_bf16_kernel<<<(n2 + 255) / 256, 256, 0, stream>>>(W_attn, Wat, EE, N3E);
    int n3 = EE * EE;
    cvt_transpose_bf16_kernel<<<(n3 + 255) / 256, 256, 0, stream>>>(W_proj, Wpt, EE, EE);
  }

  // 2) QKV projection -> head-major Q/K and transposed V
  gemm_bf16_kernel<0><<<dim3(N3E / 64, MTOT / 64), 128, 0, stream>>>(
      xb, Wat, b_attn, Qb, Kb, Vt, MTOT, N3E, EE);

  // 3) causal flash attention
  attn_kernel<<<dim3(TT / 64, BB * HH), 128, 0, stream>>>(Qb, Kb, Vt, Yb);

  // 4) output projection (fp32 out)
  gemm_bf16_kernel<1><<<dim3(EE / 64, MTOT / 64), 128, 0, stream>>>(
      Yb, Wpt, b_proj, out, nullptr, nullptr, MTOT, EE, EE);
}
